// MPNEncoder_4578435138045
// MI455X (gfx1250) — matrix operand order repaired and checked by reading, not compiled, not run on hardware
//
#include <hip/hip_runtime.h>
#include <stdint.h>

#define ATOM_FDIM 133
#define BOND_FDIM 14
#define DIM       512
#define DEPTH     4
#define MAXNB     6

// padded widths (multiples of 32 so every GEMM K-loop is check-free)
#define KP_ATOM   160   // 133 -> 160
#define KP_CAT    544   // 512 + 14 -> 544
#define KP_DIM    512
#define KP_OUT    1024

typedef __bf16 bf16_t;
typedef __bf16 v16bf __attribute__((ext_vector_type(16)));
typedef __bf16 v8bf  __attribute__((ext_vector_type(8)));
typedef float  v8f   __attribute__((ext_vector_type(8)));
typedef int    v4i   __attribute__((ext_vector_type(4)));

#if __has_builtin(__builtin_amdgcn_global_load_async_to_lds_b128) && \
    __has_builtin(__builtin_amdgcn_s_wait_asynccnt)
#define HAVE_ASYNC_LDS 1
typedef __attribute__((address_space(1))) v4i* gv4i_p;   // global int4*
typedef __attribute__((address_space(3))) v4i* lv4i_p;   // LDS int4*
#else
#define HAVE_ASYNC_LDS 0
#endif

__device__ __forceinline__ bf16_t f2bf(float f) {
  uint32_t u = __builtin_bit_cast(uint32_t, f);
  u += 0x7fffu + ((u >> 16) & 1u);
  return __builtin_bit_cast(bf16_t, (uint16_t)(u >> 16));
}

// pack two fp32 -> two bf16 (round half-up) in one dword: 2 adds + 1 v_perm
__device__ __forceinline__ uint32_t pk2bf(float a, float b) {
  uint32_t ua = __builtin_bit_cast(uint32_t, a) + 0x8000u;
  uint32_t ub = __builtin_bit_cast(uint32_t, b) + 0x8000u;
  return __builtin_amdgcn_perm(ub, ua, 0x07060302u);  // {b.hi16, a.hi16}
}

// ---------------------------------------------------------------------------
// WMMA GEMM: out = epilogue( A[M x Kp] @ Wt^T + bias ),  N == DIM == 512
//   Wt: bf16, transposed+padded [DIM][Kp]; A rows Kp-wide zero-padded.
//   MODE 0: out = relu(v)
//   MODE 2: out = v + selfadd[row]          (in-place self-message update)
//   MODE 3: atomicAdd(pooled[molid[row]], relu(v))   (fused segment-sum)
// ---------------------------------------------------------------------------
#define BM 128
#define BN 64
#define BK 32
#define LDA 40   // padded bf16 row stride (80B)
#define LDB 40

template<int MODE>
__global__ __launch_bounds__(256)
void mpn_wmma_gemm(const float* __restrict__ A, int lda, int Kp,
                   const bf16_t* __restrict__ Wt,
                   const float* __restrict__ bias,
                   float* out, int ldo, const float* selfadd,
                   const int* __restrict__ molids, int n_mols, int M)
{
  __shared__ __align__(16) bf16_t As[2 * BM * LDA];
  __shared__ __align__(16) bf16_t Bs[2 * BN * LDB];

  const int t        = threadIdx.x;
  const int lane     = t & 31;
  const int wid      = t >> 5;
  const int wm       = wid & 3;
  const int wn       = wid >> 2;
  const int blockRow = blockIdx.x * BM;
  const int blockCol = blockIdx.y * BN;

  const int halfsel = lane >> 4;
  const int lane16  = lane & 15;
  const int kbaseA  = halfsel * 8;
  const int kbaseB  = halfsel * 8;

  const int bNN = t >> 2;            // B staging: one 16B chunk per thread
  const int bK8 = (t & 3) * 8;

  v8f acc[2][2];
  const v8f vzero = {0.f,0.f,0.f,0.f,0.f,0.f,0.f,0.f};
  acc[0][0] = vzero; acc[0][1] = vzero; acc[1][0] = vzero; acc[1][1] = vzero;

  float4 va[4];
#if !HAVE_ASYNC_LDS
  uint4  vb;
#endif

  auto loadA = [&](int kt) {
    const int k0 = kt * BK;
#pragma unroll
    for (int i = 0; i < 4; ++i) {
      int q = t + i * 256;
      int r = q >> 3, c4 = (q & 7) * 4;
      int grow = blockRow + r;
      if (grow >= M) grow = M - 1;   // clamp: valid memory, dead output rows
      va[i] = *(const float4*)(A + (size_t)grow * lda + k0 + c4);
    }
  };
  auto stageA = [&](int buf) {
    bf16_t* as = As + buf * (BM * LDA);
#pragma unroll
    for (int i = 0; i < 4; ++i) {
      int q = t + i * 256;
      int r = q >> 3, c4 = (q & 7) * 4;
      uint2 p; p.x = pk2bf(va[i].x, va[i].y); p.y = pk2bf(va[i].z, va[i].w);
      *(uint2*)&as[r * LDA + c4] = p;
    }
  };
  // B staging: prefetch (issue) + commit (make visible in LDS buffer `buf`)
  auto prefetchB = [&](int kt, int buf) {
    const bf16_t* src = Wt + (size_t)(blockCol + bNN) * Kp + kt * BK + bK8;
#if HAVE_ASYNC_LDS
    bf16_t* dst = &Bs[buf * (BN * LDB) + bNN * LDB + bK8];
    __builtin_amdgcn_global_load_async_to_lds_b128(
        (gv4i_p)(v4i*)(void*)const_cast<bf16_t*>(src),
        (lv4i_p)(v4i*)(void*)dst, 0, 0);
#else
    (void)buf;
    vb = *(const uint4*)src;
#endif
  };
  auto commitB = [&](int buf) {
#if HAVE_ASYNC_LDS
    (void)buf;
    __builtin_amdgcn_s_wait_asynccnt(0);
#else
    *(uint4*)&Bs[buf * (BN * LDB) + bNN * LDB + bK8] = vb;
#endif
  };

  const int ktiles = Kp / BK;

  prefetchB(0, 0);
  loadA(0);
  stageA(0);
  commitB(0);
  __syncthreads();

  for (int kt = 0; kt < ktiles; ++kt) {
    const int b = kt & 1;
    const bool more = (kt + 1 < ktiles);
    if (more) { loadA(kt + 1); prefetchB(kt + 1, 1 - b); }

    const bf16_t* as = As + b * (BM * LDA);
    const bf16_t* bs = Bs + b * (BN * LDB);
    v16bf afr[2], bfr[2];
#pragma unroll
    for (int sm = 0; sm < 2; ++sm) {
      int rA = wm * 32 + sm * 16 + lane16;
      v8bf lo = *(const v8bf*)&as[rA * LDA + kbaseA];
      v8bf hi = *(const v8bf*)&as[rA * LDA + 16 + kbaseA];
      afr[sm] = __builtin_shufflevector(lo, hi, 0,1,2,3,4,5,6,7,8,9,10,11,12,13,14,15);
    }
#pragma unroll
    for (int sn = 0; sn < 2; ++sn) {
      int cB = wn * 32 + sn * 16 + lane16;
      v8bf lo = *(const v8bf*)&bs[cB * LDB + kbaseB];
      v8bf hi = *(const v8bf*)&bs[cB * LDB + kbaseB + 16];
      bfr[sn] = __builtin_shufflevector(lo, hi, 0,1,2,3,4,5,6,7,8,9,10,11,12,13,14,15);
    }
#pragma unroll
    for (int sm = 0; sm < 2; ++sm)
#pragma unroll
      for (int sn = 0; sn < 2; ++sn)
        acc[sm][sn] = __builtin_amdgcn_wmma_f32_16x16x32_bf16(
            false, afr[sm], false, bfr[sn], (short)0, acc[sm][sn], false, false);

    if (more) { stageA(1 - b); commitB(1 - b); }
    __syncthreads();
  }

  // epilogue: VGPR r -> row halfsel*8 + r, col = lane16
#pragma unroll
  for (int sm = 0; sm < 2; ++sm) {
    int rbase = blockRow + wm * 32 + sm * 16 + halfsel * 8;
#pragma unroll
    for (int sn = 0; sn < 2; ++sn) {
      int col = blockCol + wn * 32 + sn * 16 + lane16;
      float bv = bias[col];
#pragma unroll
      for (int r = 0; r < 8; ++r) {
        int row = rbase + r;
        if (row >= M) continue;
        float v = acc[sm][sn][r] + bv;
        if (MODE == 0) {
          out[(size_t)row * ldo + col] = fmaxf(v, 0.f);
        } else if (MODE == 2) {
          out[(size_t)row * ldo + col] = v + selfadd[(size_t)row * DIM + col];
        } else {
          int mid = molids[row];
          if (mid < n_mols)
            __hip_atomic_fetch_add(&out[(size_t)mid * DIM + col], fmaxf(v, 0.f),
                                   __ATOMIC_RELAXED, __HIP_MEMORY_SCOPE_AGENT);
        }
      }
    }
  }
}

// ---------------------------------------------------------------------------
// nei_atom_sum: out[row][0:512] = (relu?)(sum_j S[a2a[row][j]]), skip idx 0
__global__ void mpn_atom_gather(const float* __restrict__ S,
                                const int* __restrict__ a2a,
                                float* __restrict__ out, int ldo, int doRelu,
                                int M)
{
  int tid = blockIdx.x * blockDim.x + threadIdx.x;
  if (tid >= M * (DIM / 4)) return;
  int row = tid >> 7;
  int c4  = (tid & 127) << 2;
  const int* nb = a2a + (size_t)row * MAXNB;
  float4 s = make_float4(0.f, 0.f, 0.f, 0.f);
#pragma unroll
  for (int j = 0; j < MAXNB; ++j) {
    int idx = nb[j];
    if (idx != 0) {
      float4 v = *(const float4*)(S + (size_t)idx * DIM + c4);
      s.x += v.x; s.y += v.y; s.z += v.z; s.w += v.w;
    }
  }
  if (doRelu) {
    s.x = fmaxf(s.x, 0.f); s.y = fmaxf(s.y, 0.f);
    s.z = fmaxf(s.z, 0.f); s.w = fmaxf(s.w, 0.f);
  }
  *(float4*)(out + (size_t)row * ldo + c4) = s;
}

// relu(nei_bond_sum) into cols [512,544) of the 544-wide cat buffer; tail 0
__global__ void mpn_bond_gather(const float* __restrict__ fb,
                                const int* __restrict__ a2b,
                                float* __restrict__ G, int M)
{
  int tid = blockIdx.x * blockDim.x + threadIdx.x;
  if (tid >= M * 32) return;
  int row = tid >> 5;
  int col = tid & 31;
  float s = 0.f;
  if (col < BOND_FDIM) {
    const int* nb = a2b + (size_t)row * MAXNB;
#pragma unroll
    for (int j = 0; j < MAXNB; ++j)
      s += fb[(size_t)nb[j] * BOND_FDIM + col];
    s = fmaxf(s, 0.f);
  }
  G[(size_t)row * KP_CAT + DIM + col] = s;
}

// f_atoms -> zero-padded [M,160]
__global__ void mpn_pad_atoms(const float* __restrict__ fa,
                              float* __restrict__ out, int M)
{
  int tid = blockIdx.x * blockDim.x + threadIdx.x;
  if (tid >= M * KP_ATOM) return;
  int row = tid / KP_ATOM;
  int col = tid - row * KP_ATOM;
  out[tid] = (col < ATOM_FDIM) ? fa[(size_t)row * ATOM_FDIM + col] : 0.f;
}

// W[K,512] fp32 -> Wt[512,Kp] bf16, zero-padded (once per launch)
__global__ void mpn_cvt_weight(const float* __restrict__ W,
                               bf16_t* __restrict__ Wt, int K, int Kp)
{
  int tid = blockIdx.x * blockDim.x + threadIdx.x;
  if (tid >= DIM * Kp) return;
  int n = tid / Kp;
  int k = tid - n * Kp;
  Wt[tid] = (k < K) ? f2bf(W[(size_t)k * DIM + n]) : f2bf(0.f);
}

__global__ void mpn_zero(float* p, int n)
{
  int tid = blockIdx.x * blockDim.x + threadIdx.x;
  if (tid < n) p[tid] = 0.f;
}

// ---------------------------------------------------------------------------
extern "C" void kernel_launch(void* const* d_in, const int* in_sizes, int n_in,
                              void* d_out, int out_size, void* d_ws, size_t ws_size,
                              hipStream_t stream)
{
  const float* f_atoms = (const float*)d_in[0];
  const float* f_bonds = (const float*)d_in[1];
  const int*   a2a     = (const int*)  d_in[2];
  const int*   a2b     = (const int*)  d_in[3];
  const int*   mol_ids = (const int*)  d_in[4];
  const float* W_i_w  = (const float*)d_in[6];
  const float* W_i_b  = (const float*)d_in[7];
  const float* Wh0_w  = (const float*)d_in[8];
  const float* Wh0_b  = (const float*)d_in[9];
  const float* Wh1_w  = (const float*)d_in[10];
  const float* Wh1_b  = (const float*)d_in[11];
  const float* Wh2_w  = (const float*)d_in[12];
  const float* Wh2_b  = (const float*)d_in[13];
  const float* Wah0_w = (const float*)d_in[14];
  const float* Wah0_b = (const float*)d_in[15];
  const float* Wah1_w = (const float*)d_in[16];
  const float* Wah1_b = (const float*)d_in[17];
  const float* Wah2_w = (const float*)d_in[18];
  const float* Wah2_b = (const float*)d_in[19];
  const float* W_o_w  = (const float*)d_in[20];
  const float* W_o_b  = (const float*)d_in[21];

  const int M      = in_sizes[0] / ATOM_FDIM;
  const int n_mols = out_size / DIM;
  float* pooled = (float*)d_out;

  // workspace: R [M,1024] (aliased: cat buffer stride 544 during the depth
  // loop, [cc | a_message] stride 1024 after), T [M,512] ping, S [M,512]
  // self_msg, FA [M,160] padded atoms, then bf16 transposed weights (~4MB)
  float* R  = (float*)d_ws;
  float* T  = R  + (size_t)M * KP_OUT;
  float* S  = T  + (size_t)M * DIM;
  float* FA = S  + (size_t)M * DIM;
  bf16_t* WTi  = (bf16_t*)(FA + (size_t)M * KP_ATOM);
  bf16_t* WTh0 = WTi  + (size_t)DIM * KP_ATOM;
  bf16_t* WTh1 = WTh0 + (size_t)DIM * KP_CAT;
  bf16_t* WTh2 = WTh1 + (size_t)DIM * KP_DIM;
  bf16_t* WTa0 = WTh2 + (size_t)DIM * KP_DIM;
  bf16_t* WTa1 = WTa0 + (size_t)DIM * KP_ATOM;
  bf16_t* WTa2 = WTa1 + (size_t)DIM * KP_DIM;
  bf16_t* WTo  = WTa2 + (size_t)DIM * KP_DIM;

  dim3 blk(256);
  dim3 grd((M + BM - 1) / BM, DIM / BN);
  #define CEILB(n) (((n) + 255) / 256)

  mpn_zero<<<CEILB(n_mols * DIM), 256, 0, stream>>>(pooled, n_mols * DIM);

  mpn_cvt_weight<<<CEILB(DIM * KP_ATOM), 256, 0, stream>>>(W_i_w,  WTi,  ATOM_FDIM,     KP_ATOM);
  mpn_cvt_weight<<<CEILB(DIM * KP_CAT),  256, 0, stream>>>(Wh0_w,  WTh0, DIM+BOND_FDIM, KP_CAT);
  mpn_cvt_weight<<<CEILB(DIM * KP_DIM),  256, 0, stream>>>(Wh1_w,  WTh1, DIM,           KP_DIM);
  mpn_cvt_weight<<<CEILB(DIM * KP_DIM),  256, 0, stream>>>(Wh2_w,  WTh2, DIM,           KP_DIM);
  mpn_cvt_weight<<<CEILB(DIM * KP_ATOM), 256, 0, stream>>>(Wah0_w, WTa0, ATOM_FDIM,     KP_ATOM);
  mpn_cvt_weight<<<CEILB(DIM * KP_DIM),  256, 0, stream>>>(Wah1_w, WTa1, DIM,           KP_DIM);
  mpn_cvt_weight<<<CEILB(DIM * KP_DIM),  256, 0, stream>>>(Wah2_w, WTa2, DIM,           KP_DIM);
  mpn_cvt_weight<<<CEILB(DIM * KP_OUT),  256, 0, stream>>>(W_o_w,  WTo,  2*DIM,         KP_OUT);
  mpn_pad_atoms<<<CEILB(M * KP_ATOM), 256, 0, stream>>>(f_atoms, FA, M);
  mpn_bond_gather<<<CEILB(M * 32), 256, 0, stream>>>(f_bonds, a2b, R, M);

  // S = msg = relu(f_atoms @ W_i + b), row 0 zeroed
  mpn_wmma_gemm<0><<<grd, blk, 0, stream>>>(FA, KP_ATOM, KP_ATOM, WTi, W_i_b,
                                            S, DIM, nullptr, nullptr, 0, M);
  mpn_zero<<<1, DIM, 0, stream>>>(S, DIM);

  const int ag = M * (DIM / 4);
  for (int d = 0; d < DEPTH; ++d) {
    // R[:,0:512] = relu-part of concat (relu moved out of GEMM staging)
    mpn_atom_gather<<<CEILB(ag), 256, 0, stream>>>(S, a2a, R, KP_CAT, 1, M);
    mpn_wmma_gemm<0><<<grd, blk, 0, stream>>>(R, KP_CAT, KP_CAT, WTh0, Wh0_b,
                                              T, DIM, nullptr, nullptr, 0, M);
    mpn_wmma_gemm<0><<<grd, blk, 0, stream>>>(T, KP_DIM, KP_DIM, WTh1, Wh1_b,
                                              T, DIM, nullptr, nullptr, 0, M);
    mpn_wmma_gemm<2><<<grd, blk, 0, stream>>>(T, KP_DIM, KP_DIM, WTh2, Wh2_b,
                                              S, DIM, S, nullptr, 0, M);
  }

  // a_message (raw, no relu) -> CCAM high half; cc chain -> CCAM low half
  mpn_atom_gather<<<CEILB(ag), 256, 0, stream>>>(S, a2a, R + DIM, KP_OUT, 0, M);
  mpn_wmma_gemm<0><<<grd, blk, 0, stream>>>(FA, KP_ATOM, KP_ATOM, WTa0, Wah0_b,
                                            T, DIM, nullptr, nullptr, 0, M);
  mpn_wmma_gemm<0><<<grd, blk, 0, stream>>>(T, KP_DIM, KP_DIM, WTa1, Wah1_b,
                                            T, DIM, nullptr, nullptr, 0, M);
  mpn_wmma_gemm<0><<<grd, blk, 0, stream>>>(T, KP_DIM, KP_DIM, WTa2, Wah2_b,
                                            R, KP_OUT, nullptr, nullptr, 0, M);

  // out = relu([cc | a_message] @ W_o + b) fused with per-molecule segment sum
  mpn_wmma_gemm<3><<<grd, blk, 0, stream>>>(R, KP_OUT, KP_OUT, WTo, W_o_b,
                                            pooled, DIM, nullptr, mol_ids,
                                            n_mols, M);
  (void)n_in; (void)ws_size;
}
